// MambaModel_78950088835616
// MI455X (gfx1250) — matrix operand order repaired and checked by reading, not compiled, not run on hardware
//
#include <hip/hip_runtime.h>
#include <hip/hip_bf16.h>
#include <math.h>

// ---------------- model constants (must match reference) ----------------
#define BSZ     8
#define SEQ     4096
#define DINPUT  64
#define DMODEL  256
#define DOUT    10
#define DINNER  512
#define NHEADS  8
#define HEADDIM 64
#define DSTATE  64
#define DCONV   4
#define CONVDIM (DINNER + 2 * DSTATE)              // 640
#define DINPROJ (2 * DINNER + 2 * DSTATE + NHEADS) // 1160
#define NLAYERS 2
#define TOK     (BSZ * SEQ)                        // 32768
#define EPSF    1e-5f

// ---------------- WMMA vector types (gfx1250, wave32) ----------------
typedef __attribute__((ext_vector_type(16))) __bf16          v16bf;
typedef __attribute__((ext_vector_type(8)))  float           v8f;
typedef __attribute__((ext_vector_type(8)))  unsigned short  v8u;
typedef __attribute__((ext_vector_type(16))) unsigned short  v16u;

static __device__ __forceinline__ unsigned short f2bf(float f) {
  // round-to-nearest-even fp32 -> bf16
  unsigned int u = __builtin_bit_cast(unsigned int, f);
  u += 0x7FFFu + ((u >> 16) & 1u);
  return (unsigned short)(u >> 16);
}

static __device__ __forceinline__ v16bf ld_frag(const unsigned short* p0,
                                                const unsigned short* p1) {
  const v8u lo = *(const v8u*)p0;
  const v8u hi = *(const v8u*)p1;
  v16u u;
#pragma unroll
  for (int i = 0; i < 8; ++i) { u[i] = lo[i]; u[i + 8] = hi[i]; }
  return __builtin_bit_cast(v16bf, u);
}

// =====================================================================
// GEMM: C[M,N] = A[M,K] @ B[K,N] (+bias[N]); fp32 in/out, bf16 multiply
// via v_wmma_f32_16x16x32_bf16, fp32 accumulate.
// Block = 256 threads (8 waves). Block tile 128(M) x 64(N), K step 32.
// Wave w owns rows [m0+16w, +16) and all four 16-wide N tiles:
//   4 accumulators, 4 static WMMAs issued back-to-back per K step.
// Tile loads are registers-first so global_load_b128s batch under one wait.
// M % 128 == 0, K % 32 == 0 required; N edge handled block-uniformly.
// =====================================================================
__global__ void __launch_bounds__(256)
gemm_bf16_wmma(const float* __restrict__ A, const float* __restrict__ B,
               const float* __restrict__ bias, float* __restrict__ C,
               int M, int N, int K)
{
  __shared__ __align__(16) unsigned short As[128 * 40]; // 128x32 bf16, stride 40 (pad)
  __shared__ __align__(16) unsigned short Bs[64 * 40];  // transposed Bs[n][k], stride 40

  const int tid  = threadIdx.x;
  const int wave = tid >> 5;
  const int lane = tid & 31;
  const int m0   = blockIdx.y * 128;
  const int n0   = blockIdx.x * 64;
  const bool fullN = (n0 + 64 <= N);     // block-uniform edge test

  v8f acc[4] = {};

  for (int kk = 0; kk < K; kk += 32) {
    // ---- issue ALL global tile loads first (batched under one wait) ----
    float4 a4[4];
#pragma unroll
    for (int i = 0; i < 4; ++i) {
      const int r = (tid >> 3) + i * 32;   // 0..127
      const int c = (tid & 7) * 4;         // 0..28
      a4[i] = *(const float4*)(A + (size_t)(m0 + r) * K + kk + c);
    }
    float4 b4[2];
    if (fullN) {
#pragma unroll
      for (int i = 0; i < 2; ++i) {
        const int r = (tid >> 4) + i * 16; // k row 0..31
        const int c = (tid & 15) * 4;      // n col 0..60
        b4[i] = *(const float4*)(B + (size_t)(kk + r) * N + n0 + c);
      }
    } else {
#pragma unroll
      for (int i = 0; i < 2; ++i) {
        const int r = (tid >> 4) + i * 16;
        const int c = (tid & 15) * 4;
        const float* src = B + (size_t)(kk + r) * N + n0 + c;
        b4[i].x = (n0 + c + 0 < N) ? src[0] : 0.f;
        b4[i].y = (n0 + c + 1 < N) ? src[1] : 0.f;
        b4[i].z = (n0 + c + 2 < N) ? src[2] : 0.f;
        b4[i].w = (n0 + c + 3 < N) ? src[3] : 0.f;
      }
    }

    // ---- convert + store to LDS ----
#pragma unroll
    for (int i = 0; i < 4; ++i) {
      const int r = (tid >> 3) + i * 32;
      const int c = (tid & 7) * 4;
      unsigned short* d = &As[r * 40 + c];
      d[0] = f2bf(a4[i].x); d[1] = f2bf(a4[i].y);
      d[2] = f2bf(a4[i].z); d[3] = f2bf(a4[i].w);
    }
#pragma unroll
    for (int i = 0; i < 2; ++i) {
      const int r = (tid >> 4) + i * 16;
      const int c = (tid & 15) * 4;
      Bs[(c + 0) * 40 + r] = f2bf(b4[i].x);
      Bs[(c + 1) * 40 + r] = f2bf(b4[i].y);
      Bs[(c + 2) * 40 + r] = f2bf(b4[i].z);
      Bs[(c + 3) * 40 + r] = f2bf(b4[i].w);
    }
    __syncthreads();

    // ---- WMMA: preload all fragments, then 4 back-to-back WMMAs ----
    {
      const int nl   = lane & 15;
      const int mrow = wave * 16 + nl;
      const int kbA  = (lane >> 4) * 8;     // 0 or 8  (16-bit A lane layout)
      const int kbB  = (lane >> 4) * 8;     // 0 or 8  (16-bit B lane layout)
      const v16bf av = ld_frag(&As[mrow * 40 + kbA], &As[mrow * 40 + kbA + 16]);
      v16bf bv[4];
#pragma unroll
      for (int j = 0; j < 4; ++j) {
        const int ncol = j * 16 + nl;
        bv[j] = ld_frag(&Bs[ncol * 40 + kbB], &Bs[ncol * 40 + kbB + 16]);
      }
#pragma unroll
      for (int j = 0; j < 4; ++j)
        acc[j] = __builtin_amdgcn_wmma_f32_16x16x32_bf16(
            false, av, false, bv[j], (short)0, acc[j], false, false);
    }
    __syncthreads();
  }

  // ---- store: VGPR r -> row rbase + r; col = n tile + (lane & 15) ----
  const int nl    = lane & 15;
  const int rbase = m0 + wave * 16 + ((lane >> 4) ? 8 : 0);
  if (fullN) {
#pragma unroll
    for (int j = 0; j < 4; ++j) {
      const int ncol = n0 + j * 16 + nl;
      const float bs = bias ? bias[ncol] : 0.f;
#pragma unroll
      for (int r = 0; r < 8; ++r)
        C[(size_t)(rbase + r) * N + ncol] = acc[j][r] + bs;
    }
  } else {
#pragma unroll
    for (int j = 0; j < 4; ++j) {
      const int ncol = n0 + j * 16 + nl;
      if (ncol < N) {
        const float bs = bias ? bias[ncol] : 0.f;
#pragma unroll
        for (int r = 0; r < 8; ++r)
          C[(size_t)(rbase + r) * N + ncol] = acc[j][r] + bs;
      }
    }
  }
}

// =====================================================================
// Depthwise causal conv (D_CONV=4) over time + SiLU, on xBC slice of zxbcdt.
// =====================================================================
__global__ void __launch_bounds__(256)
conv_silu_kernel(const float* __restrict__ zxbcdt, const float* __restrict__ conv_w,
                 const float* __restrict__ conv_b, float* __restrict__ xBC)
{
  const int idx = blockIdx.x * 256 + threadIdx.x;
  if (idx >= TOK * CONVDIM) return;
  const int c  = idx % CONVDIM;
  const int bt = idx / CONVDIM;
  const int t  = bt % SEQ;
  const float* w = conv_w + c * DCONV;
  float acc = conv_b[c];
#pragma unroll
  for (int j = 0; j < DCONV; ++j) {
    const int tt = t - (DCONV - 1) + j;
    if (tt >= 0)
      acc += w[j] * zxbcdt[(size_t)(bt - (DCONV - 1) + j) * DINPROJ + DINNER + c];
  }
  xBC[(size_t)bt * CONVDIM + c] = acc / (1.f + __expf(-acc));  // silu
}

// =====================================================================
// dt = softplus(raw + dt_bias); dA = exp(dt * (-exp(A_log)))
// =====================================================================
__global__ void __launch_bounds__(256)
dt_kernel(const float* __restrict__ zxbcdt, const float* __restrict__ dt_bias,
          const float* __restrict__ A_log, float* __restrict__ dtb,
          float* __restrict__ dAb)
{
  const int idx = blockIdx.x * 256 + threadIdx.x;
  if (idx >= TOK * NHEADS) return;
  const int h  = idx % NHEADS;
  const int bt = idx / NHEADS;
  const float x  = zxbcdt[(size_t)bt * DINPROJ + DINNER + CONVDIM + h] + dt_bias[h];
  const float sp = (x > 20.f) ? x : log1pf(__expf(x));
  const float Ah = -__expf(A_log[h]);
  dtb[idx] = sp;
  dAb[idx] = __expf(sp * Ah);
}

// =====================================================================
// Sequential SSD scan. One block per (batch, head); 256 threads.
// Thread t owns p = t>>2 (head dim) and 16 states n = (t&3)*16 .. +15.
// y[b,t,h,p] = sum_n h[p,n]*C[n] + Dp[h]*x[p]
// =====================================================================
__global__ void __launch_bounds__(256)
scan_kernel(const float* __restrict__ xBC, const float* __restrict__ dtb,
            const float* __restrict__ dAb, const float* __restrict__ Dp,
            float* __restrict__ yscan)
{
  const int b   = blockIdx.x / NHEADS;
  const int h   = blockIdx.x % NHEADS;
  const int tid = threadIdx.x;
  const int p   = tid >> 2;
  const int ng  = tid & 3;

  __shared__ float sx[HEADDIM];
  __shared__ float sB[DSTATE];
  __shared__ float sC[DSTATE];
  __shared__ float sdt[2];

  float st[16];
#pragma unroll
  for (int i = 0; i < 16; ++i) st[i] = 0.f;
  const float Dh = Dp[h];
  const size_t rowbase = (size_t)b * SEQ;

  for (int t = 0; t < SEQ; ++t) {
    const size_t row = rowbase + t;
    const float* xrow = xBC + row * CONVDIM;
    if (tid < 64)        sx[tid]       = xrow[h * HEADDIM + tid];
    else if (tid < 128)  sB[tid - 64]  = xrow[DINNER + (tid - 64)];
    else if (tid < 192)  sC[tid - 128] = xrow[DINNER + DSTATE + (tid - 128)];
    else if (tid == 192) sdt[0] = dtb[row * NHEADS + h];
    else if (tid == 193) sdt[1] = dAb[row * NHEADS + h];
    if (tid == 194 && t + 1 < SEQ)
      __builtin_prefetch(xrow + CONVDIM + h * HEADDIM, 0, 1); // global_prefetch_b8
    __syncthreads();

    const float xv = sx[p];
    const float coeff = sdt[0] * xv;
    const float dAv = sdt[1];
    float part = 0.f;
#pragma unroll
    for (int i = 0; i < 16; ++i) {
      const int n = ng * 16 + i;
      st[i] = st[i] * dAv + coeff * sB[n];
      part += st[i] * sC[n];
    }
    part += __shfl_down(part, 1, 4);
    part += __shfl_down(part, 2, 4);
    if (ng == 0)
      yscan[row * DINNER + h * HEADDIM + p] = part + Dh * xv;
    __syncthreads();
  }
}

// =====================================================================
// y = y * silu(z); y = y * rsqrt(mean(y^2)+eps) * norm_w  (in-place on y)
// =====================================================================
__global__ void __launch_bounds__(256)
gate_rmsnorm_kernel(float* __restrict__ y, const float* __restrict__ zxbcdt,
                    const float* __restrict__ norm_w)
{
  const int row = blockIdx.x;
  const int tid = threadIdx.x;
  __shared__ float red[256];
  float v[2];
  float ss = 0.f;
#pragma unroll
  for (int i = 0; i < 2; ++i) {
    const int c = tid + i * 256;
    const float z  = zxbcdt[(size_t)row * DINPROJ + c];
    const float yv = y[(size_t)row * DINNER + c] * (z / (1.f + __expf(-z)));
    v[i] = yv;
    ss += yv * yv;
  }
  red[tid] = ss;
  __syncthreads();
  for (int s = 128; s > 0; s >>= 1) { if (tid < s) red[tid] += red[tid + s]; __syncthreads(); }
  const float scale = rsqrtf(red[0] / (float)DINNER + EPSF);
#pragma unroll
  for (int i = 0; i < 2; ++i) {
    const int c = tid + i * 256;
    y[(size_t)row * DINNER + c] = v[i] * scale * norm_w[c];
  }
}

// =====================================================================
// h = LayerNorm(h + y_proj) over DMODEL=256, in-place on H.
// =====================================================================
__global__ void __launch_bounds__(256)
add_ln_kernel(float* __restrict__ H, const float* __restrict__ ytmp,
              const float* __restrict__ ln_w, const float* __restrict__ ln_b)
{
  const int row = blockIdx.x;
  const int tid = threadIdx.x;
  __shared__ float red[256];
  const float hv = H[(size_t)row * DMODEL + tid] + ytmp[(size_t)row * DMODEL + tid];
  red[tid] = hv;
  __syncthreads();
  for (int s = 128; s > 0; s >>= 1) { if (tid < s) red[tid] += red[tid + s]; __syncthreads(); }
  const float mu = red[0] / (float)DMODEL;
  __syncthreads();
  const float d = hv - mu;
  red[tid] = d * d;
  __syncthreads();
  for (int s = 128; s > 0; s >>= 1) { if (tid < s) red[tid] += red[tid + s]; __syncthreads(); }
  const float var = red[0] / (float)DMODEL;
  H[(size_t)row * DMODEL + tid] = d * rsqrtf(var + EPSF) * ln_w[tid] + ln_b[tid];
}

// =====================================================================
// out[b,o] = H[b, L-1, :] @ dec_w + dec_b  (tiny: 8x10)
// =====================================================================
__global__ void __launch_bounds__(128)
dec_kernel(const float* __restrict__ H, const float* __restrict__ dec_w,
           const float* __restrict__ dec_b, float* __restrict__ out)
{
  const int idx = blockIdx.x * 128 + threadIdx.x;
  if (idx >= BSZ * DOUT) return;
  const int o = idx % DOUT;
  const int b = idx / DOUT;
  const float* hrow = H + ((size_t)b * SEQ + SEQ - 1) * DMODEL;
  float acc = dec_b[o];
  for (int d = 0; d < DMODEL; ++d) acc += hrow[d] * dec_w[d * DOUT + o];
  out[idx] = acc;
}

// =====================================================================
// Host-side orchestration (graph-capture safe: only launches on stream).
// =====================================================================
extern "C" void kernel_launch(void* const* d_in, const int* in_sizes, int n_in,
                              void* d_out, int out_size, void* d_ws, size_t ws_size,
                              hipStream_t stream)
{
  const float* x         = (const float*)d_in[0];   // (8,4096,64)
  const float* enc_w     = (const float*)d_in[1];   // (64,256)
  const float* enc_b     = (const float*)d_in[2];   // (256)
  const float* in_proj_w = (const float*)d_in[3];   // (2,256,1160)
  const float* conv_w    = (const float*)d_in[4];   // (2,640,4)
  const float* conv_b    = (const float*)d_in[5];   // (2,640)
  const float* dt_bias   = (const float*)d_in[6];   // (2,8)
  const float* A_log     = (const float*)d_in[7];   // (2,8)
  const float* Dp        = (const float*)d_in[8];   // (2,8)
  const float* norm_w    = (const float*)d_in[9];   // (2,512)
  const float* out_pw    = (const float*)d_in[10];  // (2,512,256)
  const float* ln_w      = (const float*)d_in[11];  // (2,256)
  const float* ln_b      = (const float*)d_in[12];  // (2,256)
  const float* dec_w     = (const float*)d_in[13];  // (256,10)
  const float* dec_b     = (const float*)d_in[14];  // (10)
  float* out = (float*)d_out;                       // (8,10)

  // workspace carve-up (floats); total ~93.1M floats ~= 372 MB
  float* ws = (float*)d_ws;
  float* H      = ws;                      ws += (size_t)TOK * DMODEL;
  float* zx     = ws;                      ws += (size_t)TOK * DINPROJ;
  float* xBC    = ws;                      ws += (size_t)TOK * CONVDIM;
  float* dtb    = ws;                      ws += (size_t)TOK * NHEADS;
  float* dAb    = ws;                      ws += (size_t)TOK * NHEADS;
  float* yscan  = ws;                      ws += (size_t)TOK * DINNER;
  float* ytmp   = ws;                      ws += (size_t)TOK * DMODEL;
  (void)ws_size; (void)in_sizes; (void)n_in; (void)out_size;

  const dim3 blk256(256), blk128(128);

  // 1) encoder: H = x @ enc_w + enc_b
  gemm_bf16_wmma<<<dim3(DMODEL / 64, TOK / 128), blk256, 0, stream>>>(
      x, enc_w, enc_b, H, TOK, DMODEL, DINPUT);

  for (int layer = 0; layer < NLAYERS; ++layer) {
    const float* ipw = in_proj_w + (size_t)layer * DMODEL * DINPROJ;
    const float* cw  = conv_w   + (size_t)layer * CONVDIM * DCONV;
    const float* cb  = conv_b   + (size_t)layer * CONVDIM;
    const float* dtbs= dt_bias  + (size_t)layer * NHEADS;
    const float* al  = A_log    + (size_t)layer * NHEADS;
    const float* dp  = Dp       + (size_t)layer * NHEADS;
    const float* nw  = norm_w   + (size_t)layer * DINNER;
    const float* opw = out_pw   + (size_t)layer * DINNER * DMODEL;
    const float* lw  = ln_w     + (size_t)layer * DMODEL;
    const float* lb  = ln_b     + (size_t)layer * DMODEL;

    // 2a) in-projection: zx = H @ in_proj_w   (N=1160: last block is edge)
    gemm_bf16_wmma<<<dim3((DINPROJ + 63) / 64, TOK / 128), blk256, 0, stream>>>(
        H, ipw, nullptr, zx, TOK, DINPROJ, DMODEL);

    // 2b) causal depthwise conv + SiLU on xBC slice
    conv_silu_kernel<<<dim3((TOK * CONVDIM + 255) / 256), blk256, 0, stream>>>(
        zx, cw, cb, xBC);

    // 2c) dt softplus + dA
    dt_kernel<<<dim3((TOK * NHEADS + 255) / 256), blk256, 0, stream>>>(
        zx, dtbs, al, dtb, dAb);

    // 2d) SSD scan (+ Dp*x skip term)
    scan_kernel<<<dim3(BSZ * NHEADS), blk256, 0, stream>>>(
        xBC, dtb, dAb, dp, yscan);

    // 2e) gate with silu(z) + RMSNorm (in-place on yscan)
    gate_rmsnorm_kernel<<<dim3(TOK), blk256, 0, stream>>>(yscan, zx, nw);

    // 2f) out-projection: ytmp = yscan @ out_proj_w
    gemm_bf16_wmma<<<dim3(DMODEL / 64, TOK / 128), blk256, 0, stream>>>(
        yscan, opw, nullptr, ytmp, TOK, DMODEL, DINNER);

    // 2g) residual add + LayerNorm (in-place on H)
    add_ln_kernel<<<dim3(TOK), blk256, 0, stream>>>(H, ytmp, lw, lb);
  }

  // 3) decoder on last token
  dec_kernel<<<dim3(1), blk128, 0, stream>>>(H, dec_w, dec_b, out);
}
